// PitchEnergyPredictor_49349174231633
// MI455X (gfx1250) — matrix operand order repaired and checked by reading, not compiled, not run on hardware
//
#include <hip/hip_runtime.h>
#include <hip/hip_bf16.h>
#include <math.h>

// ---------------- problem constants ----------------
#define L_     8
#define E_     640
#define DHID_  512
#define STYLE_ 128
#define ID_    1280
#define NH_    4
#define DH_    320
#define NB_    4
#define BS_    320
#define B_     2
#define T_     1024
#define INTER_ 1024
#define SCALE_ 0.05590169943749474f   // 1/sqrt(320)

typedef __attribute__((ext_vector_type(16))) __bf16 bf16x16;
typedef __attribute__((ext_vector_type(8)))  float  f32x8;

static __device__ __forceinline__ float siluf(float x){ return x / (1.f + __expf(-x)); }
static __device__ __forceinline__ float geluf(float x){ return 0.5f*x*(1.f+erff(x*0.70710678118f)); }

static __device__ __forceinline__ float block_reduce(float v, float* red, int nthr) {
  int tid = threadIdx.x;
  red[tid] = v; __syncthreads();
  for (int o = nthr >> 1; o > 0; o >>= 1) { if (tid < o) red[tid] += red[tid + o]; __syncthreads(); }
  float r = red[0]; __syncthreads();
  return r;
}

// -------- CDNA5 async global->LDS copies (ASYNCcnt-tracked) --------
static __device__ __forceinline__ void async_b128(unsigned lds_off, const float* g) {
  unsigned long long ga = (unsigned long long)g;
  asm volatile("global_load_async_to_lds_b128 %0, %1, off"
               :: "v"(lds_off), "v"(ga) : "memory");
}
static __device__ __forceinline__ void async_b32(unsigned lds_off, const float* g) {
  unsigned long long ga = (unsigned long long)g;
  asm volatile("global_load_async_to_lds_b32 %0, %1, off"
               :: "v"(lds_off), "v"(ga) : "memory");
}
static __device__ __forceinline__ void wait_async0() {
  asm volatile("s_wait_asynccnt 0" ::: "memory");
}

// -------- fp32(LDS) -> bf16 register fragments --------
static __device__ __forceinline__ void cvt8(const float* p, bf16x16& f, int base) {
  float4 x0 = *(const float4*)p;
  float4 x1 = *(const float4*)(p + 4);
  f[base+0]=(__bf16)x0.x; f[base+1]=(__bf16)x0.y; f[base+2]=(__bf16)x0.z; f[base+3]=(__bf16)x0.w;
  f[base+4]=(__bf16)x1.x; f[base+5]=(__bf16)x1.y; f[base+6]=(__bf16)x1.z; f[base+7]=(__bf16)x1.w;
}
static __device__ __forceinline__ bf16x16 cvt8x2(const float* p0, const float* p1) {
  bf16x16 f; cvt8(p0, f, 0); cvt8(p1, f, 8); return f;
}
static __device__ __forceinline__ bf16x16 cvt16(const float* p) {
  bf16x16 f; cvt8(p, f, 0); cvt8(p + 8, f, 8); return f;
}

// ---------------- generic bf16 WMMA GEMM ----------------
// C[z] = epilogue(A[z] (MxK) @ W[z] (KxN or NxK if TRANSB) + bias, act) [+ Res] [causal zero]
#define BM 128
#define BN 64
#define BK 32
#define LDT 36               // fp32 LDS tile row stride (36*4B = 144B, 16B-aligned)
#define GF_TRANSB 1
#define GF_CAUSAL 2
#define ACT_NONE 0
#define ACT_SILU 1
#define ACT_GELU 2

__global__ __launch_bounds__(256)
void gemm_bf16_wmma(const float* __restrict__ A, long strideA, int lda,
                    const float* __restrict__ Wt, long strideW, int ldb,
                    const float* __restrict__ bias, long strideBias,
                    const float* __restrict__ Res, long strideR, int ldr,
                    float* __restrict__ C, long strideC, int ldc,
                    int M, int N, int K, int act, int flags)
{
  __shared__ float Asf[2][BM * LDT];   // 128 rows (m) x 32 (k), fp32
  __shared__ float Bsf[2][BN * LDT];   // 64 rows (n) x 32 (k), fp32 (n-major!)
  (void)M; (void)N;
  int z = blockIdx.z;
  A  += (long)z * strideA;
  Wt += (long)z * strideW;
  C  += (long)z * strideC;
  if (bias) bias += (long)z * strideBias;
  if (Res)  Res  += (long)z * strideR;

  int tid = threadIdx.x, lane = tid & 31, wave = tid >> 5;
  int gm0 = blockIdx.y * BM;
  int gn0 = blockIdx.x * BN;
  int wm = (wave >> 1) * 32;   // 4 waves in M
  int wn = (wave & 1) * 32;    // 2 waves in N

  unsigned aBase[2] = { (unsigned)(unsigned long long)&Asf[0][0],
                        (unsigned)(unsigned long long)&Asf[1][0] };
  unsigned bBase[2] = { (unsigned)(unsigned long long)&Bsf[0][0],
                        (unsigned)(unsigned long long)&Bsf[1][0] };

  f32x8 acc[2][2] = {};
  int nk = K / BK;

  auto issue = [&](int it, int buf) {
    int kt = it * BK;
    // A tile: 128x32 = 1024 float4 slots, 4 per thread (async b128 -> LDS)
    #pragma unroll
    for (int i = 0; i < 4; ++i) {
      int idx = tid + i * 256;
      int r = idx >> 3, c4 = (idx & 7) * 4;
      async_b128(aBase[buf] + (unsigned)(r * LDT + c4) * 4u,
                 A + (long)(gm0 + r) * lda + kt + c4);
    }
    if (flags & GF_TRANSB) {
      // W is N x K row-major: rows map directly to n-major LDS tile
      #pragma unroll
      for (int i = 0; i < 2; ++i) {
        int idx = tid + i * 256;          // 64 n x 8 float4(k)
        int n = idx >> 3, k4 = (idx & 7) * 4;
        async_b128(bBase[buf] + (unsigned)(n * LDT + k4) * 4u,
                   Wt + (long)(gn0 + n) * ldb + kt + k4);
      }
    } else {
      // W is K x N row-major: scatter k-rows into n-major tile (b32, lane-coalesced over n)
      int n = tid & 63;
      int k0 = (tid >> 6) * 8;
      #pragma unroll
      for (int j = 0; j < 8; ++j) {
        int k = k0 + j;
        async_b32(bBase[buf] + (unsigned)(n * LDT + k) * 4u,
                  Wt + (long)(kt + k) * ldb + gn0 + n);
      }
    }
  };

  issue(0, 0);
  for (int it = 0; it < nk; ++it) {
    int cur = it & 1;
    wait_async0();          // my async writes for tile 'it' are in LDS
    __syncthreads();        // everyone's are; everyone finished reading buf cur^1
    if (it + 1 < nk) issue(it + 1, cur ^ 1);

    bf16x16 afrag[2], bfrag[2];
    // A fragment (ISA 16-bit A 16x32 layout): two contiguous 8-float runs per lane
    int kbA = (lane & 16) ? 8 : 0;
    #pragma unroll
    for (int ti = 0; ti < 2; ++ti) {
      int m = wm + ti * 16 + (lane & 15);
      afrag[ti] = cvt8x2(&Asf[cur][m * LDT + kbA], &Asf[cur][m * LDT + kbA + 16]);
    }
    // B fragment (ISA 16-bit B 32x16 layout): two contiguous 8-float runs per lane
    int kbB = (lane & 16) ? 8 : 0;
    #pragma unroll
    for (int tj = 0; tj < 2; ++tj) {
      int n = wn + tj * 16 + (lane & 15);
      bfrag[tj] = cvt8x2(&Bsf[cur][n * LDT + kbB], &Bsf[cur][n * LDT + kbB + 16]);
    }
    #pragma unroll
    for (int ti = 0; ti < 2; ++ti)
      #pragma unroll
      for (int tj = 0; tj < 2; ++tj)
        acc[ti][tj] = __builtin_amdgcn_wmma_f32_16x16x32_bf16(
            false, afrag[ti], false, bfrag[tj], (short)0, acc[ti][tj], false, false);
  }

  // epilogue (C/D layout: VGPR r -> M = r (+8 for lanes 16-31), N = lane&15)
  #pragma unroll
  for (int ti = 0; ti < 2; ++ti) {
    #pragma unroll
    for (int tj = 0; tj < 2; ++tj) {
      int n = gn0 + wn + tj * 16 + (lane & 15);
      int mbase = gm0 + wm + ti * 16 + ((lane & 16) ? 8 : 0);
      float bval = bias ? bias[n] : 0.f;
      #pragma unroll
      for (int r = 0; r < 8; ++r) {
        int m = mbase + r;
        float v = acc[ti][tj][r] + bval;
        if (act == ACT_SILU) v = siluf(v);
        else if (act == ACT_GELU) v = geluf(v);
        if (Res) v += Res[(long)m * ldr + n];
        if ((flags & GF_CAUSAL) && n > m) v = 0.f;
        C[(long)m * ldc + n] = v;
      }
    }
  }
}

// ---------------- small kernels ----------------
__global__ __launch_bounds__(256)
void transpose_in(const float* __restrict__ P, float* __restrict__ X) {
  long idx = (long)blockIdx.x * 256 + threadIdx.x;
  if (idx >= (long)B_ * T_ * E_) return;
  int e = idx % E_; long bt = idx / E_; int t = bt % T_; long b = bt / T_;
  X[idx] = P[(b * E_ + e) * T_ + t];
}

__global__ __launch_bounds__(256)
void layernorm_rows(const float* __restrict__ X, float* __restrict__ Y,
                    const float* __restrict__ w, const float* __restrict__ b, int F)
{
  long row = blockIdx.x;
  const float* x = X + row * F;
  float* y = Y + row * F;
  __shared__ float red[256];
  float s = 0;
  for (int i = threadIdx.x; i < F; i += 256) s += x[i];
  float mu = block_reduce(s, red, 256) / F;
  float v = 0;
  for (int i = threadIdx.x; i < F; i += 256) { float d = x[i] - mu; v += d * d; }
  float rs = rsqrtf(block_reduce(v, red, 256) / F + 1e-5f);
  for (int i = threadIdx.x; i < F; i += 256)
    y[i] = (x[i] - mu) * rs * w[i] + b[i];
}

// causal depthwise conv K=4 + SiLU; xm = first ID cols of UP (b,s,2ID)
__global__ __launch_bounds__(256)
void dwconv4_silu(const float* __restrict__ UP, const float* __restrict__ cw,
                  const float* __restrict__ cb, float* __restrict__ XA)
{
  long idx = (long)blockIdx.x * 256 + threadIdx.x;
  if (idx >= (long)B_ * T_ * ID_) return;
  int c = idx % ID_;
  long bs = idx / ID_;
  int s = bs % T_; long b = bs / T_;
  float acc = cb[c];
  #pragma unroll
  for (int j = 0; j < 4; ++j) {
    int t = s - 3 + j;
    if (t >= 0) acc += UP[(b * T_ + t) * (long)(2 * ID_) + c] * cw[c * 4 + j];
  }
  XA[idx] = siluf(acc);
}

// ig/fg gate GEMV: qkv = concat(q,k,v); outputs (b,h,s)
__global__ __launch_bounds__(256)
void gates_kernel(const float* __restrict__ Q, const float* __restrict__ Kb,
                  const float* __restrict__ V,
                  const float* __restrict__ igw, const float* __restrict__ igb,
                  const float* __restrict__ fgw, const float* __restrict__ fgb,
                  float* __restrict__ IG, float* __restrict__ FG)
{
  long bs = blockIdx.x;
  int s = bs % T_; long b = bs / T_;
  const float* q = Q + bs * ID_;
  const float* k = Kb + bs * ID_;
  const float* v = V + bs * ID_;
  __shared__ float red[256];
  float aI[NH_] = {0, 0, 0, 0}, aF[NH_] = {0, 0, 0, 0};
  for (int i = threadIdx.x; i < ID_; i += 256) {
    float qv = q[i], kv = k[i], vv = v[i];
    #pragma unroll
    for (int h = 0; h < NH_; ++h) {
      aI[h] += qv * igw[(long)i * NH_ + h] + kv * igw[(long)(ID_ + i) * NH_ + h] + vv * igw[(long)(2 * ID_ + i) * NH_ + h];
      aF[h] += qv * fgw[(long)i * NH_ + h] + kv * fgw[(long)(ID_ + i) * NH_ + h] + vv * fgw[(long)(2 * ID_ + i) * NH_ + h];
    }
  }
  for (int h = 0; h < NH_; ++h) {
    float ri = block_reduce(aI[h], red, 256);
    if (threadIdx.x == 0) IG[(b * NH_ + h) * (long)T_ + s] = ri + igb[h];
    float rf = block_reduce(aF[h], red, 256);
    if (threadIdx.x == 0) FG[(b * NH_ + h) * (long)T_ + s] = rf + fgb[h];
  }
}

// per (b,h) scan: lc = cumsum(log_sigmoid(fg)); m_s = lc_s + prefixmax(ig-lc)
__global__ void decay_kernel(const float* __restrict__ IG, const float* __restrict__ FG,
                             float* __restrict__ ROWS, float* __restrict__ COLS,
                             float* __restrict__ ENM)
{
  int z = blockIdx.x * blockDim.x + threadIdx.x;
  if (z >= B_ * NH_) return;
  const float* ig = IG + (long)z * T_;
  const float* fg = FG + (long)z * T_;
  float lc = 0.f, pm = -3.0e38f;
  for (int s = 0; s < T_; ++s) {
    float f = fg[s];
    float ls = (f > 0.f) ? -log1pf(expf(-f)) : (f - log1pf(expf(f)));
    lc += ls;
    float a = ig[s] - lc;
    pm = fmaxf(pm, a);
    float m = lc + pm;
    ROWS[(long)z * T_ + s] = expf(lc - m) * SCALE_;  // row factor * SCALE
    COLS[(long)z * T_ + s] = expf(ig[s] - lc);       // column factor (bf16-range safe)
    ENM[(long)z * T_ + s]  = expf(-m);
  }
}

// pack (b,s,ID) head-blocks into (b,h,s,DH) with decay scaling folded in
__global__ __launch_bounds__(256)
void pack_scale_qkv(const float* __restrict__ Q, const float* __restrict__ Kb,
                    const float* __restrict__ V,
                    const float* __restrict__ ROWS, const float* __restrict__ COLS,
                    float* __restrict__ QS, float* __restrict__ KS, float* __restrict__ VP)
{
  long idx = (long)blockIdx.x * 256 + threadIdx.x;
  if (idx >= (long)B_ * NH_ * T_ * DH_) return;
  int d = idx % DH_;
  long r = idx / DH_;
  int s = r % T_;
  long zh = r / T_;
  int h = zh % NH_;
  long b = zh / NH_;
  long src = (b * T_ + s) * (long)ID_ + h * DH_ + d;
  QS[idx] = Q[src] * ROWS[zh * T_ + s];
  KS[idx] = Kb[src] * COLS[zh * T_ + s];
  VP[idx] = V[src];
}

// n_s = max(|sum_t C|, exp(-m_s)) + 1e-6 ; C /= n_s
__global__ __launch_bounds__(256)
void attn_norm_rows(float* __restrict__ CM, const float* __restrict__ ENM)
{
  long row = blockIdx.x;               // (b*NH+h)*T + s
  float* c = CM + row * (long)T_;
  int s = row % T_;
  __shared__ float red[256];
  float sum = 0;
  for (int t = threadIdx.x; t <= s; t += 256) sum += c[t];
  float tot = block_reduce(sum, red, 256);
  float inv = 1.f / (fmaxf(fabsf(tot), ENM[row]) + 1e-6f);
  for (int t = threadIdx.x; t < T_; t += 256) c[t] *= inv;
}

// head-norm + mhn_w + skip*xa, then * silu(z); writes merged (b,s,ID)
__global__ __launch_bounds__(128)
void headnorm_combine(const float* __restrict__ HC, const float* __restrict__ XA,
                      const float* __restrict__ UP, const float* __restrict__ mhn,
                      const float* __restrict__ skip, float* __restrict__ G)
{
  long r = blockIdx.x;                 // (b*NH+h)*T + s
  int s = r % T_;
  long zh = r / T_;
  int h = zh % NH_;
  long b = zh / NH_;
  const float* hc = HC + r * DH_;
  __shared__ float red[128];
  float sum = 0;
  for (int d = threadIdx.x; d < DH_; d += 128) sum += hc[d];
  float mu = block_reduce(sum, red, 128) / DH_;
  float v = 0;
  for (int d = threadIdx.x; d < DH_; d += 128) { float dd = hc[d] - mu; v += dd * dd; }
  float rs = rsqrtf(block_reduce(v, red, 128) / DH_ + 1e-5f);
  long bt = b * T_ + s;
  for (int d = threadIdx.x; d < DH_; d += 128) {
    long o = bt * ID_ + h * DH_ + d;
    float hn = (hc[d] - mu) * rs * mhn[h * DH_ + d];
    float ht = hn + skip[h * DH_ + d] * XA[o];
    float zz = UP[bt * (long)(2 * ID_) + ID_ + h * DH_ + d];
    G[o] = ht * siluf(zz);
  }
}

// ada: out[h][b][:] = style[b] @ ada_w[h] + ada_b[h]
__global__ __launch_bounds__(256)
void ada_kernel(const float* __restrict__ style, const float* __restrict__ aw,
                const float* __restrict__ ab, float* __restrict__ out, int cin2)
{
  int b = blockIdx.x, h = blockIdx.y;
  const float* s = style + (long)b * STYLE_;
  const float* w = aw + (long)h * STYLE_ * cin2;
  for (int o = threadIdx.x; o < cin2; o += 256) {
    float acc = ab[(long)h * cin2 + o];
    for (int i = 0; i < STYLE_; ++i) acc += s[i] * w[(long)i * cin2 + o];
    out[((long)h * B_ + b) * cin2 + o] = acc;
  }
}

// depthwise conv7 (pad 3,3) + channel-LN + AdaLN modulation; layout (h,b,t,c)
__global__ __launch_bounds__(256)
void cnx_dw_ln_mod(const float* __restrict__ Xin, long strideInH, int cin,
                   const float* __restrict__ dw, const float* __restrict__ db,
                   const float* __restrict__ ada, float* __restrict__ OUT)
{
  int t = blockIdx.x, b = blockIdx.y, h = blockIdx.z;
  const float* x = Xin + (long)h * strideInH + (long)b * T_ * cin;
  __shared__ float vals[512];
  __shared__ float red[256];
  const float* dwh = dw + (long)h * cin * 7;
  const float* dbh = db + (long)h * cin;
  for (int c = threadIdx.x; c < cin; c += 256) {
    float acc = dbh[c];
    #pragma unroll
    for (int j = 0; j < 7; ++j) {
      int tt = t - 3 + j;
      if (tt >= 0 && tt < T_) acc += x[(long)tt * cin + c] * dwh[c * 7 + j];
    }
    vals[c] = acc;
  }
  __syncthreads();
  float s = 0;
  for (int c = threadIdx.x; c < cin; c += 256) s += vals[c];
  float mu = block_reduce(s, red, 256) / cin;
  float v = 0;
  for (int c = threadIdx.x; c < cin; c += 256) { float d = vals[c] - mu; v += d * d; }
  float rs = rsqrtf(block_reduce(v, red, 256) / cin + 1e-5f);
  const float* g  = ada + ((long)h * B_ + b) * 2 * cin;
  const float* be = g + cin;
  float* out = OUT + (((long)h * B_ + b) * T_ + t) * (long)cin;
  for (int c = threadIdx.x; c < cin; c += 256)
    out[c] = (1.f + g[c]) * (vals[c] - mu) * rs + be[c];
}

// out[h,b,t] = y3[h,b,t,:] . proj_w[h] + proj_b[h]
__global__ __launch_bounds__(256)
void final_proj(const float* __restrict__ Y3, const float* __restrict__ pw,
                const float* __restrict__ pb, float* __restrict__ OUT)
{
  long r = blockIdx.x;                 // h*B*T + b*T + t
  int hh = (int)(r / ((long)B_ * T_));
  const float* y = Y3 + r * 256;
  __shared__ float red[256];
  float v = y[threadIdx.x] * pw[hh * 256 + threadIdx.x];
  float tot = block_reduce(v, red, 256);
  if (threadIdx.x == 0) OUT[r] = tot + pb[hh];
}

// ---------------- host side ----------------
static void launch_gemm(hipStream_t s, const float* A, long sA, int lda,
                        const float* W, long sW, int ldb,
                        const float* bias, long sBias,
                        const float* Res, long sR, int ldr,
                        float* C, long sC, int ldc,
                        int M, int N, int K, int batch, int act, int flags)
{
  dim3 grid(N / BN, M / BM, batch);
  gemm_bf16_wmma<<<grid, 256, 0, s>>>(A, sA, lda, W, sW, ldb, bias, sBias,
                                      Res, sR, ldr, C, sC, ldc, M, N, K, act, flags);
}

extern "C" void kernel_launch(void* const* d_in, const int* in_sizes, int n_in,
                              void* d_out, int out_size, void* d_ws, size_t ws_size,
                              hipStream_t stream)
{
  (void)in_sizes; (void)n_in; (void)out_size; (void)ws_size;
  const float* prosody = (const float*)d_in[0];
  const float* style   = (const float*)d_in[1];
  const float* ln_w  = (const float*)d_in[2];
  const float* ln_b  = (const float*)d_in[3];
  const float* Wup   = (const float*)d_in[4];
  const float* conv_w = (const float*)d_in[5];
  const float* conv_b = (const float*)d_in[6];
  const float* Wq = (const float*)d_in[7];
  const float* Wk = (const float*)d_in[8];
  const float* Wv = (const float*)d_in[9];
  const float* ig_w = (const float*)d_in[10];
  const float* ig_b = (const float*)d_in[11];
  const float* fg_w = (const float*)d_in[12];
  const float* fg_b = (const float*)d_in[13];
  const float* skip = (const float*)d_in[14];
  const float* mhn_w = (const float*)d_in[15];
  const float* Wdown = (const float*)d_in[16];
  const float* out_ln_w = (const float*)d_in[17];
  const float* out_ln_b = (const float*)d_in[18];
  const float* prep_W = (const float*)d_in[19];
  const float* prep_b = (const float*)d_in[20];
  const float* dw_w[3]  = {(const float*)d_in[21], (const float*)d_in[29], (const float*)d_in[37]};
  const float* dw_b[3]  = {(const float*)d_in[22], (const float*)d_in[30], (const float*)d_in[38]};
  const float* ada_w[3] = {(const float*)d_in[23], (const float*)d_in[31], (const float*)d_in[39]};
  const float* ada_b[3] = {(const float*)d_in[24], (const float*)d_in[32], (const float*)d_in[40]};
  const float* pa_w[3]  = {(const float*)d_in[25], (const float*)d_in[33], (const float*)d_in[41]};
  const float* pa_b[3]  = {(const float*)d_in[26], (const float*)d_in[34], (const float*)d_in[42]};
  const float* pb_w[3]  = {(const float*)d_in[27], (const float*)d_in[35], (const float*)d_in[43]};
  const float* pb_b[3]  = {(const float*)d_in[28], (const float*)d_in[36], (const float*)d_in[44]};
  const float* res2_w = (const float*)d_in[45];
  const float* res2_b = (const float*)d_in[46];
  const float* proj_w = (const float*)d_in[47];
  const float* proj_b = (const float*)d_in[48];
  float* out = (float*)d_out;

  float* ws = (float*)d_ws;
  size_t off = 0;
  auto alloc = [&](size_t n) { float* p = ws + off; off += (n + 63) & ~(size_t)63; return p; };

  const long BT = (long)B_ * T_;
  float* X   = alloc(BT * E_);
  float* H   = alloc(BT * E_);
  float* UP  = alloc(BT * 2 * ID_);
  float* XA  = alloc(BT * ID_);
  float* Qb  = alloc(BT * ID_);
  float* Kb  = alloc(BT * ID_);
  float* Vb  = alloc(BT * ID_);
  float* IG  = alloc((long)B_ * NH_ * T_);
  float* FG  = alloc((long)B_ * NH_ * T_);
  float* RO  = alloc((long)B_ * NH_ * T_);
  float* CO  = alloc((long)B_ * NH_ * T_);
  float* ENM = alloc((long)B_ * NH_ * T_);
  float* QS  = alloc((long)B_ * NH_ * T_ * DH_);
  float* KS  = alloc((long)B_ * NH_ * T_ * DH_);
  float* VP  = alloc((long)B_ * NH_ * T_ * DH_);
  float* CM  = alloc((long)B_ * NH_ * T_ * T_);
  float* HC  = alloc((long)B_ * NH_ * T_ * DH_);
  float* GB  = alloc(BT * ID_);
  float* XP  = alloc(BT * DHID_);
  float* ADA = alloc((long)2 * B_ * 2 * INTER_);
  float* CT  = alloc((long)2 * BT * DHID_);
  float* IN2 = alloc((long)2 * BT * INTER_);
  float* Y1  = alloc((long)2 * BT * DHID_);
  float* R2  = alloc((long)2 * BT * 256);
  float* Y2  = alloc((long)2 * BT * 256);
  float* Y3  = alloc((long)2 * BT * 256);

  { long n = BT * E_; transpose_in<<<dim3((unsigned)((n + 255) / 256)), 256, 0, stream>>>(prosody, X); }

  for (int l = 0; l < L_; ++l) {
    const float* lw  = ln_w + (long)l * E_;
    const float* lb  = ln_b + (long)l * E_;
    const float* wup = Wup + (long)l * E_ * 2 * ID_;
    const float* cw  = conv_w + (long)l * ID_ * 4;
    const float* cb  = conv_b + (long)l * ID_;
    const float* wq  = Wq + (long)l * NB_ * BS_ * BS_;
    const float* wk  = Wk + (long)l * NB_ * BS_ * BS_;
    const float* wv  = Wv + (long)l * NB_ * BS_ * BS_;
    const float* igw = ig_w + (long)l * 3 * ID_ * NH_;
    const float* igb = ig_b + (long)l * NH_;
    const float* fgw = fg_w + (long)l * 3 * ID_ * NH_;
    const float* fgb = fg_b + (long)l * NH_;
    const float* sk  = skip + (long)l * ID_;
    const float* mh  = mhn_w + (long)l * ID_;
    const float* wd  = Wdown + (long)l * ID_ * E_;

    layernorm_rows<<<dim3((unsigned)BT), 256, 0, stream>>>(X, H, lw, lb, E_);
    launch_gemm(stream, H, 0, E_, wup, 0, 2 * ID_, nullptr, 0, nullptr, 0, 0,
                UP, 0, 2 * ID_, (int)BT, 2 * ID_, E_, 1, ACT_NONE, 0);
    { long n = BT * ID_; dwconv4_silu<<<dim3((unsigned)((n + 255) / 256)), 256, 0, stream>>>(UP, cw, cb, XA); }
    // block-diagonal headwise projections (batch over NB blocks via strideA=BS)
    launch_gemm(stream, XA, BS_, ID_, wq, (long)BS_ * BS_, BS_, nullptr, 0, nullptr, 0, 0,
                Qb, BS_, ID_, (int)BT, BS_, BS_, NB_, ACT_NONE, 0);
    launch_gemm(stream, XA, BS_, ID_, wk, (long)BS_ * BS_, BS_, nullptr, 0, nullptr, 0, 0,
                Kb, BS_, ID_, (int)BT, BS_, BS_, NB_, ACT_NONE, 0);
    launch_gemm(stream, UP, BS_, 2 * ID_, wv, (long)BS_ * BS_, BS_, nullptr, 0, nullptr, 0, 0,
                Vb, BS_, ID_, (int)BT, BS_, BS_, NB_, ACT_NONE, 0);
    gates_kernel<<<dim3((unsigned)BT), 256, 0, stream>>>(Qb, Kb, Vb, igw, igb, fgw, fgb, IG, FG);
    decay_kernel<<<1, 32, 0, stream>>>(IG, FG, RO, CO, ENM);
    { long n = (long)B_ * NH_ * T_ * DH_;
      pack_scale_qkv<<<dim3((unsigned)((n + 255) / 256)), 256, 0, stream>>>(Qb, Kb, Vb, RO, CO, QS, KS, VP); }
    // C = qs @ ks^T, causal zeroed
    launch_gemm(stream, QS, (long)T_ * DH_, DH_, KS, (long)T_ * DH_, DH_, nullptr, 0, nullptr, 0, 0,
                CM, (long)T_ * T_, T_, T_, T_, DH_, B_ * NH_, ACT_NONE, GF_TRANSB | GF_CAUSAL);
    attn_norm_rows<<<dim3((unsigned)(B_ * NH_ * T_)), 256, 0, stream>>>(CM, ENM);
    // hc = Cn @ v
    launch_gemm(stream, CM, (long)T_ * T_, T_, VP, (long)T_ * DH_, DH_, nullptr, 0, nullptr, 0, 0,
                HC, (long)T_ * DH_, DH_, T_, DH_, T_, B_ * NH_, ACT_NONE, 0);
    headnorm_combine<<<dim3((unsigned)(B_ * NH_ * T_)), 128, 0, stream>>>(HC, XA, UP, mh, sk, GB);
    // x = x + g @ Wdown (fused residual)
    launch_gemm(stream, GB, 0, ID_, wd, 0, E_, nullptr, 0, X, 0, E_,
                X, 0, E_, (int)BT, E_, ID_, 1, ACT_NONE, 0);
  }

  layernorm_rows<<<dim3((unsigned)BT), 256, 0, stream>>>(X, H, out_ln_w, out_ln_b, E_);
  launch_gemm(stream, H, 0, E_, prep_W, 0, DHID_, prep_b, 0, nullptr, 0, 0,
              XP, 0, DHID_, (int)BT, DHID_, E_, 1, ACT_NONE, 0);

  // ---- ConvNeXt head stage 1 (cin=512 -> 512), input shared across heads ----
  ada_kernel<<<dim3(B_, 2), 256, 0, stream>>>(style, ada_w[0], ada_b[0], ADA, 2 * 512);
  cnx_dw_ln_mod<<<dim3(T_, B_, 2), 256, 0, stream>>>(XP, 0, 512, dw_w[0], dw_b[0], ADA, CT);
  launch_gemm(stream, CT, BT * 512, 512, pa_w[0], (long)512 * INTER_, INTER_, pa_b[0], INTER_,
              nullptr, 0, 0, IN2, BT * INTER_, INTER_, (int)BT, INTER_, 512, 2, ACT_GELU, 0);
  launch_gemm(stream, IN2, BT * INTER_, INTER_, pb_w[0], (long)INTER_ * 512, 512, pb_b[0], 512,
              XP, 0, 512, Y1, BT * 512, 512, (int)BT, 512, INTER_, 2, ACT_NONE, 0);
  // ---- stage 2 (512 -> 256 with residual projection) ----
  launch_gemm(stream, Y1, BT * 512, 512, res2_w, (long)512 * 256, 256, res2_b, 256,
              nullptr, 0, 0, R2, BT * 256, 256, (int)BT, 256, 512, 2, ACT_NONE, 0);
  ada_kernel<<<dim3(B_, 2), 256, 0, stream>>>(style, ada_w[1], ada_b[1], ADA, 2 * 512);
  cnx_dw_ln_mod<<<dim3(T_, B_, 2), 256, 0, stream>>>(Y1, BT * 512, 512, dw_w[1], dw_b[1], ADA, CT);
  launch_gemm(stream, CT, BT * 512, 512, pa_w[1], (long)512 * INTER_, INTER_, pa_b[1], INTER_,
              nullptr, 0, 0, IN2, BT * INTER_, INTER_, (int)BT, INTER_, 512, 2, ACT_GELU, 0);
  launch_gemm(stream, IN2, BT * INTER_, INTER_, pb_w[1], (long)INTER_ * 256, 256, pb_b[1], 256,
              R2, BT * 256, 256, Y2, BT * 256, 256, (int)BT, 256, INTER_, 2, ACT_NONE, 0);
  // ---- stage 3 (256 -> 256) ----
  ada_kernel<<<dim3(B_, 2), 256, 0, stream>>>(style, ada_w[2], ada_b[2], ADA, 2 * 256);
  cnx_dw_ln_mod<<<dim3(T_, B_, 2), 256, 0, stream>>>(Y2, BT * 256, 256, dw_w[2], dw_b[2], ADA, CT);
  launch_gemm(stream, CT, BT * 256, 256, pa_w[2], (long)256 * INTER_, INTER_, pa_b[2], INTER_,
              nullptr, 0, 0, IN2, BT * INTER_, INTER_, (int)BT, INTER_, 256, 2, ACT_GELU, 0);
  launch_gemm(stream, IN2, BT * INTER_, INTER_, pb_w[2], (long)INTER_ * 256, 256, pb_b[2], 256,
              Y2, BT * 256, 256, Y3, BT * 256, 256, (int)BT, 256, INTER_, 2, ACT_NONE, 0);

  final_proj<<<dim3((unsigned)(2 * B_ * T_)), 256, 0, stream>>>(Y3, proj_w, proj_b, out);
}
